// Language_70574902608316
// MI455X (gfx1250) — matrix operand order repaired and checked by reading, not compiled, not run on hardware
//
#include <hip/hip_runtime.h>
#include <hip/hip_bf16.h>

// ---------------------------------------------------------------------------
// Problem constants (from reference)
// ---------------------------------------------------------------------------
#define BB   256      // batch
#define TT   64       // seq len
#define VV   6207     // vocab
#define DD   512      // embedding dim
#define HH   512      // hidden
#define YF   2048     // image feature dim
#define G4   2048     // 4*H
#define EPSC 1e-10f

typedef __bf16 bf16_t;
typedef __attribute__((ext_vector_type(16))) __bf16 v16bf;
typedef __attribute__((ext_vector_type(8)))  __bf16 v8bf;
typedef __attribute__((ext_vector_type(8)))  float  v8f;

// ---------------------------------------------------------------------------
// WMMA fragment helpers (CDNA5 wave32 layouts, CDNA5 ISA §7.12.2)
// ---------------------------------------------------------------------------
// A 16x32 bf16: lane m = lane&15, khi = lane>>4; lane holds K = khi*16 .. +15
//   (16 bf16, 32 bytes contiguous; same k order as B)
__device__ __forceinline__ v16bf load_a_frag(const bf16_t* __restrict__ row_k0, int khi) {
    return *(const v16bf*)(row_k0 + khi * 16);
}

// B 32x16 bf16: lane n = lane&15, khi = lane>>4; lane holds K = khi*16 .. +15
// B[k][n] = W[n][k] (weights row-major N x K) -> 32 contiguous bytes per lane.
__device__ __forceinline__ v16bf load_b_frag(const bf16_t* __restrict__ W, int nrow,
                                             int ldb, int k0, int khi) {
    return *(const v16bf*)(W + (size_t)nrow * ldb + k0 + khi * 16);
}

__device__ __forceinline__ v8f wmma_bf16(v16bf a, v16bf b, v8f c) {
    return __builtin_amdgcn_wmma_f32_16x16x32_bf16(false, a, false, b, (short)0, c,
                                                   false, false);
}

__device__ __forceinline__ float sigmf(float x) { return 1.0f / (1.0f + __expf(-x)); }

// ---------------------------------------------------------------------------
// Utility kernels
// ---------------------------------------------------------------------------
__global__ void f32_to_bf16_kernel(const float* __restrict__ src,
                                   bf16_t* __restrict__ dst, int n) {
    for (int i = blockIdx.x * blockDim.x + threadIdx.x; i < n;
         i += gridDim.x * blockDim.x)
        dst[i] = (bf16_t)src[i];
}

__global__ void bias_sum_kernel(const float* __restrict__ a, const float* __restrict__ b,
                                float* __restrict__ o, int n) {
    int i = blockIdx.x * blockDim.x + threadIdx.x;
    if (i < n) o[i] = a[i] + b[i];
}

__global__ void zero_f32_kernel(float* __restrict__ p, int n) {
    int i = blockIdx.x * blockDim.x + threadIdx.x;
    if (i < n) p[i] = 0.0f;
}

// ---------------------------------------------------------------------------
// Fully fused LSTM cell (GEMMs + gate nonlinearity in one kernel):
//   z = A1 @ W1^T + A2 @ W2^T + bsum          (pre-activations, i|f|g|o)
//   c' = sig(f)*c + sig(i)*tanh(g) ;  h' = sig(o)*tanh(c')
//
// N-tiling owns a 32-column slice of ALL FOUR gates per workgroup so the
// cell update can complete in-kernel:
//   8 waves = 4 gates x 2 M-halves; wave tile 16(M) x 32(N) of one gate.
//   Pre-activation tiles exchanged through LDS (padded vs bank aliasing),
//   then 256 threads finish the elementwise update for the 32x32 h/c tile.
// A1 optionally gathered from the embedding table (fused lookup).
// Grid: (HH/32, BB/32) = (16, 8).
// ---------------------------------------------------------------------------
__global__ __launch_bounds__(256)
void lstm_fused_kernel(const bf16_t* __restrict__ A1, const bf16_t* __restrict__ emb,
                       const int* __restrict__ x, int t,
                       const bf16_t* __restrict__ A2,
                       const bf16_t* __restrict__ W1, const bf16_t* __restrict__ W2,
                       const float* __restrict__ bsum,
                       float* __restrict__ C, float* __restrict__ H,
                       bf16_t* __restrict__ Hb) {
    __shared__ float zs[4][32][33];          // [gate][m][col], +1 pad
    const int lane = threadIdx.x & 31;
    const int wave = threadIdx.x >> 5;
    const int gate = wave & 3;               // 0:i 1:f 2:g 3:o
    const int wm   = wave >> 2;              // M half
    const int m0   = blockIdx.y * 32 + wm * 16;
    const int c0   = blockIdx.x * 32;        // column slice within H
    const int n0   = gate * HH + c0;         // column in 4H pre-activation space
    const int ml = lane & 15, khi = lane >> 4;

    const int arow = m0 + ml;
    const bf16_t* a1row = x ? (emb + (size_t)x[arow * TT + t] * DD)
                            : (A1 + (size_t)arow * DD);
    const bf16_t* a2row = A2 + (size_t)arow * HH;

    v8f acc0 = {}; v8f acc1 = {};
#pragma unroll 2
    for (int k0 = 0; k0 < DD; k0 += 32) {
        v16bf a  = load_a_frag(a1row + k0, khi);
        v16bf b0 = load_b_frag(W1, n0 + ml,      DD, k0, khi);
        v16bf b1 = load_b_frag(W1, n0 + 16 + ml, DD, k0, khi);
        acc0 = wmma_bf16(a, b0, acc0);
        acc1 = wmma_bf16(a, b1, acc1);
    }
#pragma unroll 2
    for (int k0 = 0; k0 < HH; k0 += 32) {
        v16bf a  = load_a_frag(a2row + k0, khi);
        v16bf b0 = load_b_frag(W2, n0 + ml,      HH, k0, khi);
        v16bf b1 = load_b_frag(W2, n0 + 16 + ml, HH, k0, khi);
        acc0 = wmma_bf16(a, b0, acc0);
        acc1 = wmma_bf16(a, b1, acc1);
    }
    // D layout: n = lane&15, m = r + 8*(lane>>4); stage z tile (+bias) in LDS
    const int n = lane & 15;
#pragma unroll
    for (int r = 0; r < 8; ++r) {
        const int mm = wm * 16 + khi * 8 + r;            // 0..31 local row
        zs[gate][mm][n]      = acc0[r] + bsum[n0 + n];
        zs[gate][mm][n + 16] = acc1[r] + bsum[n0 + 16 + n];
    }
    __syncthreads();

    // Elementwise cell update for the 32(M) x 32(col) tile: 4 elems/thread
#pragma unroll
    for (int e = threadIdx.x; e < 1024; e += 256) {
        const int m = e >> 5, cc_ = e & 31;
        const int row = blockIdx.y * 32 + m;
        const size_t idx = (size_t)row * HH + c0 + cc_;
        const float zi = zs[0][m][cc_], zf = zs[1][m][cc_];
        const float zg = zs[2][m][cc_], zo = zs[3][m][cc_];
        const float cnew = sigmf(zf) * C[idx] + sigmf(zi) * tanhf(zg);
        const float hnew = sigmf(zo) * tanhf(cnew);
        C[idx] = cnew;
        H[idx] = hnew;
        Hb[idx] = (bf16_t)hnew;
    }
}

// ---------------------------------------------------------------------------
// y-gate precompute (hoisted out of the time loop):
//   ypart = y @ Wg[:,512:]^T + bg     M=256, N=512, K=2048
// Grid: (512/128, 256/32) = (4, 8).
// ---------------------------------------------------------------------------
__global__ __launch_bounds__(256)
void ygate_gemm_kernel(const bf16_t* __restrict__ Yb, const bf16_t* __restrict__ Wg2,
                       const float* __restrict__ bg, float* __restrict__ ypart) {
    const int lane = threadIdx.x & 31;
    const int wave = threadIdx.x >> 5;
    const int m0 = blockIdx.y * 32 + (wave >> 2) * 16;
    const int n0 = blockIdx.x * 128 + (wave & 3) * 32;
    const int ml = lane & 15, khi = lane >> 4;
    const bf16_t* arow = Yb + (size_t)(m0 + ml) * YF;

    v8f acc0 = {}; v8f acc1 = {};
#pragma unroll 2
    for (int k0 = 0; k0 < YF; k0 += 32) {
        v16bf a  = load_a_frag(arow + k0, khi);
        v16bf b0 = load_b_frag(Wg2, n0 + ml,      HH + YF, k0, khi);
        v16bf b1 = load_b_frag(Wg2, n0 + 16 + ml, HH + YF, k0, khi);
        acc0 = wmma_bf16(a, b0, acc0);
        acc1 = wmma_bf16(a, b1, acc1);
    }
    const int n = lane & 15;
#pragma unroll
    for (int r = 0; r < 8; ++r) {
        const int mm = m0 + khi * 8 + r;
        const int c0 = n0 + n, c1 = n0 + 16 + n;
        ypart[(size_t)mm * HH + c0] = acc0[r] + bg[c0];
        ypart[(size_t)mm * HH + c1] = acc1[r] + bg[c1];
    }
}

// ---------------------------------------------------------------------------
// Gate GEMM + gumbel-sigmoid hard gate, fused:
//   logits = relu(h @ Wg[:,:512]^T + ypart)
//   hf     = (logits + noise >= 0) ? h : 0       (straight-through forward)
//   M=256, N=512, K=512.  Grid (4, 8).
// ---------------------------------------------------------------------------
__global__ __launch_bounds__(256)
void gate_gemm_kernel(const bf16_t* __restrict__ Hb, const bf16_t* __restrict__ Wg1,
                      const float* __restrict__ ypart, const float* __restrict__ Hf,
                      const float* __restrict__ u1, const float* __restrict__ u2,
                      int t, bf16_t* __restrict__ HFb) {
    const int lane = threadIdx.x & 31;
    const int wave = threadIdx.x >> 5;
    const int m0 = blockIdx.y * 32 + (wave >> 2) * 16;
    const int n0 = blockIdx.x * 128 + (wave & 3) * 32;
    const int ml = lane & 15, khi = lane >> 4;
    const bf16_t* arow = Hb + (size_t)(m0 + ml) * HH;

    v8f acc0 = {}; v8f acc1 = {};
#pragma unroll 2
    for (int k0 = 0; k0 < HH; k0 += 32) {
        v16bf a  = load_a_frag(arow + k0, khi);
        v16bf b0 = load_b_frag(Wg1, n0 + ml,      HH + YF, k0, khi);
        v16bf b1 = load_b_frag(Wg1, n0 + 16 + ml, HH + YF, k0, khi);
        acc0 = wmma_bf16(a, b0, acc0);
        acc1 = wmma_bf16(a, b1, acc1);
    }
    const int n = lane & 15;
    const size_t ub = (size_t)t * (BB * HH);
#pragma unroll
    for (int r = 0; r < 8; ++r) {
        const int mm = m0 + khi * 8 + r;
#pragma unroll
        for (int half = 0; half < 2; ++half) {
            const int col = n0 + half * 16 + n;
            const float av = half ? acc1[r] : acc0[r];
            const size_t o = (size_t)mm * HH + col;
            float logit = fmaxf(av + ypart[o], 0.0f);
            const float l1 = __logf(u1[ub + o] + EPSC);
            const float l2 = __logf(u2[ub + o] + EPSC);
            const float noise = -__logf(l2 / l1 + EPSC);
            const float g = (logit + noise >= 0.0f) ? 1.0f : 0.0f;
            HFb[o] = (bf16_t)(Hf[o] * g);
        }
    }
}

// ---------------------------------------------------------------------------
// Final linear: out = relu(ht @ Wlin^T + blin)   M=256, N=256, K=512.
// Grid (2, 8).
// ---------------------------------------------------------------------------
__global__ __launch_bounds__(256)
void final_gemm_kernel(const bf16_t* __restrict__ HTb, const bf16_t* __restrict__ Wl,
                       const float* __restrict__ blin, float* __restrict__ out) {
    const int lane = threadIdx.x & 31;
    const int wave = threadIdx.x >> 5;
    const int m0 = blockIdx.y * 32 + (wave >> 2) * 16;
    const int n0 = blockIdx.x * 128 + (wave & 3) * 32;
    const int ml = lane & 15, khi = lane >> 4;
    const bf16_t* arow = HTb + (size_t)(m0 + ml) * HH;

    v8f acc0 = {}; v8f acc1 = {};
#pragma unroll 2
    for (int k0 = 0; k0 < HH; k0 += 32) {
        v16bf a  = load_a_frag(arow + k0, khi);
        v16bf b0 = load_b_frag(Wl, n0 + ml,      HH, k0, khi);
        v16bf b1 = load_b_frag(Wl, n0 + 16 + ml, HH, k0, khi);
        acc0 = wmma_bf16(a, b0, acc0);
        acc1 = wmma_bf16(a, b1, acc1);
    }
    const int n = lane & 15;
#pragma unroll
    for (int r = 0; r < 8; ++r) {
        const int mm = m0 + khi * 8 + r;
        const int c0 = n0 + n, c1 = n0 + 16 + n;
        out[(size_t)mm * 256 + c0] = fmaxf(acc0[r] + blin[c0], 0.0f);
        out[(size_t)mm * 256 + c1] = fmaxf(acc1[r] + blin[c1], 0.0f);
    }
}

// ---------------------------------------------------------------------------
// Host launcher
// ---------------------------------------------------------------------------
static inline size_t align256(size_t x) { return (x + 255) & ~(size_t)255; }

extern "C" void kernel_launch(void* const* d_in, const int* in_sizes, int n_in,
                              void* d_out, int out_size, void* d_ws, size_t ws_size,
                              hipStream_t stream) {
    (void)in_sizes; (void)n_in; (void)out_size; (void)ws_size;
    const float* y     = (const float*)d_in[0];
    const int*   x     = (const int*)d_in[1];
    const float* emb_w = (const float*)d_in[2];
    const float* W_ih  = (const float*)d_in[3];
    const float* W_hh  = (const float*)d_in[4];
    const float* b_ih  = (const float*)d_in[5];
    const float* b_hh  = (const float*)d_in[6];
    const float* W_iht = (const float*)d_in[7];
    const float* W_hht = (const float*)d_in[8];
    const float* b_iht = (const float*)d_in[9];
    const float* b_hht = (const float*)d_in[10];
    const float* Wg    = (const float*)d_in[11];
    const float* bg    = (const float*)d_in[12];
    const float* Wlin  = (const float*)d_in[13];
    const float* blin  = (const float*)d_in[14];
    const float* u1    = (const float*)d_in[15];
    const float* u2    = (const float*)d_in[16];
    float* out = (float*)d_out;

    // Workspace carve-up
    char* w = (char*)d_ws;
    size_t off = 0;
    auto carve = [&](size_t bytes) { void* p = w + off; off = align256(off + bytes); return p; };
    bf16_t* emb_b = (bf16_t*)carve((size_t)VV * DD * 2);
    bf16_t* W1b   = (bf16_t*)carve((size_t)G4 * DD * 2);
    bf16_t* W2b   = (bf16_t*)carve((size_t)G4 * HH * 2);
    bf16_t* W1tb  = (bf16_t*)carve((size_t)G4 * HH * 2);
    bf16_t* W2tb  = (bf16_t*)carve((size_t)G4 * HH * 2);
    bf16_t* Wgb   = (bf16_t*)carve((size_t)HH * (HH + YF) * 2);
    bf16_t* Wlb   = (bf16_t*)carve((size_t)256 * HH * 2);
    bf16_t* Yb    = (bf16_t*)carve((size_t)BB * YF * 2);
    float* bsum1  = (float*)carve(G4 * 4);
    float* bsum2  = (float*)carve(G4 * 4);
    float* ypart  = (float*)carve((size_t)BB * HH * 4);
    float* Hbot   = (float*)carve((size_t)BB * HH * 4);
    float* Cbot   = (float*)carve((size_t)BB * HH * 4);
    float* Htop   = (float*)carve((size_t)BB * HH * 4);
    float* Ctop   = (float*)carve((size_t)BB * HH * 4);
    bf16_t* Hb    = (bf16_t*)carve((size_t)BB * HH * 2);
    bf16_t* HFb   = (bf16_t*)carve((size_t)BB * HH * 2);
    bf16_t* HTb   = (bf16_t*)carve((size_t)BB * HH * 2);

    // --- weight / feature conversion to bf16 (every call; deterministic) ---
    auto cvt = [&](const float* s, bf16_t* d, size_t n) {
        int blocks = (int)((n + 255) / 256); if (blocks > 2048) blocks = 2048;
        f32_to_bf16_kernel<<<blocks, 256, 0, stream>>>(s, d, (int)n);
    };
    cvt(emb_w, emb_b, (size_t)VV * DD);
    cvt(W_ih,  W1b,  (size_t)G4 * DD);
    cvt(W_hh,  W2b,  (size_t)G4 * HH);
    cvt(W_iht, W1tb, (size_t)G4 * HH);
    cvt(W_hht, W2tb, (size_t)G4 * HH);
    cvt(Wg,    Wgb,  (size_t)HH * (HH + YF));
    cvt(Wlin,  Wlb,  (size_t)256 * HH);
    cvt(y,     Yb,   (size_t)BB * YF);
    bias_sum_kernel<<<G4 / 256, 256, 0, stream>>>(b_ih, b_hh, bsum1, G4);
    bias_sum_kernel<<<G4 / 256, 256, 0, stream>>>(b_iht, b_hht, bsum2, G4);

    // --- zero initial state (h, c, ht, ct and bf16 mirrors) ---
    const int NSTATE = BB * HH;                       // 131072 floats
    zero_f32_kernel<<<NSTATE / 256, 256, 0, stream>>>(Hbot, NSTATE);
    zero_f32_kernel<<<NSTATE / 256, 256, 0, stream>>>(Cbot, NSTATE);
    zero_f32_kernel<<<NSTATE / 256, 256, 0, stream>>>(Htop, NSTATE);
    zero_f32_kernel<<<NSTATE / 256, 256, 0, stream>>>(Ctop, NSTATE);
    zero_f32_kernel<<<(NSTATE / 2) / 256, 256, 0, stream>>>((float*)Hb,  NSTATE / 2);
    zero_f32_kernel<<<(NSTATE / 2) / 256, 256, 0, stream>>>((float*)HTb, NSTATE / 2);

    // --- hoisted y-gate partial:  ypart = y @ Wg[:,512:]^T + bg ---
    ygate_gemm_kernel<<<dim3(4, 8), 256, 0, stream>>>(Yb, Wgb + HH, bg, ypart);

    // --- recurrent time loop: 3 fused kernels per step ---
    const dim3 gCell(HH / 32, BB / 32);    // (16, 8)
    const dim3 gGate(HH / 128, BB / 32);   // (4, 8)
    for (int t = 0; t < TT; ++t) {
        // bottom cell (fused GEMMs + gates + state update, embedding gathered)
        lstm_fused_kernel<<<gCell, 256, 0, stream>>>(nullptr, emb_b, x, t, Hb,
                                                     W1b, W2b, bsum1,
                                                     Cbot, Hbot, Hb);
        // gate: hf = h * hard_gumbel(relu(h@Wg1^T + ypart) + noise)
        gate_gemm_kernel<<<gGate, 256, 0, stream>>>(Hb, Wgb, ypart, Hbot,
                                                    u1, u2, t, HFb);
        // top cell (fused)
        lstm_fused_kernel<<<gCell, 256, 0, stream>>>(HFb, emb_b, nullptr, t, HTb,
                                                     W1tb, W2tb, bsum2,
                                                     Ctop, Htop, HTb);
    }

    // --- final linear: out = relu(ht @ Wlin^T + blin) ---
    final_gemm_kernel<<<dim3(2, 8), 256, 0, stream>>>(HTb, Wlb, blin, out);
}
